// rnn_encdec_33681133535391
// MI455X (gfx1250) — matrix operand order repaired and checked by reading, not compiled, not run on hardware
//
#include <hip/hip_runtime.h>
#include <hip/hip_bf16.h>

typedef _Float16 v16h __attribute__((ext_vector_type(16)));
typedef _Float16 v8h  __attribute__((ext_vector_type(8)));
typedef float    v8f  __attribute__((ext_vector_type(8)));

#define Bz 64
#define Sz 512
#define Iz 256
#define Hz 512
#define Oz 16
#define Gz 2048   // 4*H

// ---------------------------------------------------------------------------
// WMMA operand loaders (f16, 16x16x32 shape)
// A-matrix 16x32 (MxK), row-major memory [.., K]:
//   lanes 0-15: row=lane, K chunks {0..7} in v0-3, {16..23} in v4-7
//   lanes 16-31: row=lane-16, K chunks {8..15}, {24..31}
__device__ __forceinline__ v16h load_a_tile(const _Float16* __restrict__ base,
                                            int stride, int lane) {
    const int r  = lane & 15;
    const int kb = (lane >> 4) << 3;            // 0 or 8
    const _Float16* p = base + (size_t)r * stride + kb;
    v8h lo = *reinterpret_cast<const v8h*>(p);
    v8h hi = *reinterpret_cast<const v8h*>(p + 16);
    v16h out;
#pragma unroll
    for (int i = 0; i < 8; ++i) { out[i] = lo[i]; out[i + 8] = hi[i]; }
    return out;
}

// B-matrix 32x16 (KxN). Memory holds W as [N, K] row-major (so B(k,n)=W[n,k]).
//   lanes 0-15: col=lane, K {0..7},{16..23} ; lanes 16-31: col=lane-16, K {8..15},{24..31}
__device__ __forceinline__ v16h load_b_tile(const _Float16* __restrict__ base,
                                            int stride, int lane) {
    const int n  = lane & 15;
    const int ks = (lane >> 4) << 3;            // 0 or 8
    const _Float16* p = base + (size_t)n * stride + ks;
    v8h lo = *reinterpret_cast<const v8h*>(p);
    v8h hi = *reinterpret_cast<const v8h*>(p + 16);
    v16h out;
#pragma unroll
    for (int i = 0; i < 8; ++i) { out[i] = lo[i]; out[i + 8] = hi[i]; }
    return out;
}

__device__ __forceinline__ v8f wmma_f16(v16h a, v16h b, v8f c) {
    return __builtin_amdgcn_wmma_f32_16x16x32_f16(false, a, false, b,
                                                  (short)0, c, false, false);
}

__device__ __forceinline__ float sigmoidf_(float x) {
    return 1.0f / (1.0f + __expf(-x));
}

// ---------------------------------------------------------------------------
// Generic f32 -> f16 weight convert with optional column zero-padding.
__global__ void cvt_f16_kernel(const float* __restrict__ src,
                               _Float16* __restrict__ dst,
                               int rows, int scols, int dcols) {
    int idx = blockIdx.x * blockDim.x + threadIdx.x;
    int total = rows * dcols;
    if (idx >= total) return;
    int r = idx / dcols, c = idx % dcols;
    dst[idx] = (c < scols) ? (_Float16)src[(size_t)r * scols + c] : (_Float16)0.0f;
}

// x [B,S,I] f32  ->  xh [S*B, I] f16 (time-major)
__global__ void cvt_x_kernel(const float* __restrict__ x,
                             _Float16* __restrict__ xh) {
    int idx = blockIdx.x * blockDim.x + threadIdx.x;
    if (idx >= Sz * Bz * Iz) return;
    int i = idx % Iz;
    int t = idx / Iz;
    int b = t % Bz;
    int s = t / Bz;
    xh[idx] = (_Float16)x[((size_t)b * Sz + s) * Iz + i];
}

// zero the recurrent state + feedback buffer
__global__ void init_state_kernel(_Float16* hA, _Float16* hB,
                                  float* cA, float* cB, _Float16* y) {
    int i = blockIdx.x * blockDim.x + threadIdx.x;
    if (i < Bz * Hz) {
        hA[i] = (_Float16)0.0f; hB[i] = (_Float16)0.0f;
        cA[i] = 0.0f;           cB[i] = 0.0f;
    }
    if (i < Bz * 32) y[i] = (_Float16)0.0f;
}

// ---------------------------------------------------------------------------
// Big parallel GEMM:  C[m,n] = sum_k A[m,k] * W[n,k]   (all f16, f32 accum)
__global__ void wmma_gemm_kernel(const _Float16* __restrict__ A,
                                 const _Float16* __restrict__ W,
                                 _Float16* __restrict__ C,
                                 int M, int N, int K) {
    const int wid = (blockIdx.x * blockDim.x + threadIdx.x) >> 5;
    const int ntn = N >> 4;
    if (wid >= (M >> 4) * ntn) return;
    const int lane = threadIdx.x & 31;
    const int mt = wid / ntn;
    const int nt = wid % ntn;

    v8f acc = {};
    for (int kc = 0; kc < K; kc += 32) {
        v16h a = load_a_tile(A + (size_t)(mt * 16) * K + kc, K, lane);
        v16h b = load_b_tile(W + (size_t)(nt * 16) * K + kc, K, lane);
        acc = wmma_f16(a, b, acc);
    }
    const int col = lane & 15;
    const int rb  = (lane >> 4) << 3;
#pragma unroll
    for (int r = 0; r < 8; ++r) {
        int gm = mt * 16 + rb + r;
        C[(size_t)gm * N + nt * 16 + col] = (_Float16)acc[r];
    }
}

// ---------------------------------------------------------------------------
// One LSTM recurrence step, fused gates.
// g = A1@W1^T (+ A2@W2^T) (+ Xp) + bias ;  c' = sig(f)*c + sig(i)*tanh(g)
// h' = sig(o)*tanh(c') ; h' written to h_out (f16) and optionally y_extra.
// 128 waves: mt in [0,4) (batch tiles), nt in [0,32) (H tiles); each wave
// computes the same (mt,nt) tile of all four gates.
__global__ void lstm_step_kernel(const _Float16* __restrict__ A1,
                                 const _Float16* __restrict__ W1, int K1,
                                 const _Float16* __restrict__ A2,
                                 const _Float16* __restrict__ W2, int K2,
                                 const _Float16* __restrict__ Xp,
                                 const float* __restrict__ bias,
                                 float* __restrict__ c,
                                 _Float16* __restrict__ h_out,
                                 _Float16* __restrict__ y_extra) {
    const int wid = (blockIdx.x * blockDim.x + threadIdx.x) >> 5;
    if (wid >= (Bz / 16) * (Hz / 16)) return;          // 128 waves
    const int lane = threadIdx.x & 31;
    const int mt = wid >> 5;       // Hz/16 == 32
    const int nt = wid & 31;

    v8f acc[4] = {v8f{}, v8f{}, v8f{}, v8f{}};

    for (int kc = 0; kc < K1; kc += 32) {
        v16h a = load_a_tile(A1 + (size_t)(mt * 16) * K1 + kc, K1, lane);
#pragma unroll
        for (int q = 0; q < 4; ++q) {
            v16h b = load_b_tile(W1 + (size_t)(q * Hz + nt * 16) * K1 + kc, K1, lane);
            acc[q] = wmma_f16(a, b, acc[q]);
        }
    }
    if (K2 > 0) {
        for (int kc = 0; kc < K2; kc += 32) {
            v16h a = load_a_tile(A2 + (size_t)(mt * 16) * K2 + kc, K2, lane);
#pragma unroll
            for (int q = 0; q < 4; ++q) {
                v16h b = load_b_tile(W2 + (size_t)(q * Hz + nt * 16) * K2 + kc, K2, lane);
                acc[q] = wmma_f16(a, b, acc[q]);
            }
        }
    }

    const int col = lane & 15;
    const int rb  = (lane >> 4) << 3;
    const int gh  = nt * 16 + col;
#pragma unroll
    for (int r = 0; r < 8; ++r) {
        const int gb = mt * 16 + rb + r;
        float gi = acc[0][r] + bias[0 * Hz + gh];
        float gf = acc[1][r] + bias[1 * Hz + gh];
        float gg = acc[2][r] + bias[2 * Hz + gh];
        float go = acc[3][r] + bias[3 * Hz + gh];
        if (Xp) {
            const _Float16* xp = Xp + (size_t)gb * Gz;
            gi += (float)xp[0 * Hz + gh];
            gf += (float)xp[1 * Hz + gh];
            gg += (float)xp[2 * Hz + gh];
            go += (float)xp[3 * Hz + gh];
        }
        const size_t sidx = (size_t)gb * Hz + gh;
        float cc = sigmoidf_(gf) * c[sidx] + sigmoidf_(gi) * tanhf(gg);
        float hh = sigmoidf_(go) * tanhf(cc);
        c[sidx] = cc;
        h_out[sidx] = (_Float16)hh;
        if (y_extra) y_extra[sidx] = (_Float16)hh;
    }
}

// ---------------------------------------------------------------------------
// Output projection: out[b,s,:] = h1@fcW^T + fcb ; feedback y[b,0:16]=out.
// 4 waves (one per batch tile), single N tile (O=16).
__global__ void fc_step_kernel(const _Float16* __restrict__ h1,
                               const _Float16* __restrict__ Wfc,   // [16,512] f16
                               const float* __restrict__ bfc,      // [16]
                               float* __restrict__ out,            // [B,S,O]
                               int s,
                               _Float16* __restrict__ y) {         // [B,32]
    const int wid = threadIdx.x >> 5;
    if (wid >= Bz / 16) return;
    const int lane = threadIdx.x & 31;
    const int mt = wid;

    v8f acc = {};
    for (int kc = 0; kc < Hz; kc += 32) {
        v16h a = load_a_tile(h1 + (size_t)(mt * 16) * Hz + kc, Hz, lane);
        v16h b = load_b_tile(Wfc + kc, Hz, lane);
        acc = wmma_f16(a, b, acc);
    }
    const int col = lane & 15;
    const int rb  = (lane >> 4) << 3;
#pragma unroll
    for (int r = 0; r < 8; ++r) {
        const int gb = mt * 16 + rb + r;
        float v = acc[r] + bfc[col];
        out[((size_t)gb * Sz + s) * Oz + col] = v;
        y[gb * 32 + col] = (_Float16)v;
    }
}

// ---------------------------------------------------------------------------
extern "C" void kernel_launch(void* const* d_in, const int* in_sizes, int n_in,
                              void* d_out, int out_size, void* d_ws, size_t ws_size,
                              hipStream_t stream) {
    (void)in_sizes; (void)n_in; (void)out_size; (void)ws_size;

    const float* x        = (const float*)d_in[0];
    const float* enc_Wih0 = (const float*)d_in[1];
    const float* enc_Whh0 = (const float*)d_in[2];
    const float* enc_b0   = (const float*)d_in[3];
    const float* enc_Wih1 = (const float*)d_in[4];
    const float* enc_Whh1 = (const float*)d_in[5];
    const float* enc_b1   = (const float*)d_in[6];
    const float* dec_Wih0 = (const float*)d_in[7];
    const float* dec_Whh0 = (const float*)d_in[8];
    const float* dec_b0   = (const float*)d_in[9];
    const float* dec_Wih1 = (const float*)d_in[10];
    const float* dec_Whh1 = (const float*)d_in[11];
    const float* dec_b1   = (const float*)d_in[12];
    const float* fc_W     = (const float*)d_in[13];
    const float* fc_b     = (const float*)d_in[14];
    float* out = (float*)d_out;

    // ---- workspace carve-up (f16 unless noted) ----
    size_t off = 0;
    auto alloc = [&](size_t bytes) -> char* {
        size_t cur = (off + 255) & ~(size_t)255;
        off = cur + bytes;
        return (char*)d_ws + cur;
    };
    _Float16* wih0e = (_Float16*)alloc((size_t)Gz * Iz * 2);
    _Float16* whh0e = (_Float16*)alloc((size_t)Gz * Hz * 2);
    _Float16* wih1e = (_Float16*)alloc((size_t)Gz * Hz * 2);
    _Float16* whh1e = (_Float16*)alloc((size_t)Gz * Hz * 2);
    _Float16* wih0d = (_Float16*)alloc((size_t)Gz * 32 * 2);   // padded K=16->32
    _Float16* whh0d = (_Float16*)alloc((size_t)Gz * Hz * 2);
    _Float16* wih1d = (_Float16*)alloc((size_t)Gz * Hz * 2);
    _Float16* whh1d = (_Float16*)alloc((size_t)Gz * Hz * 2);
    _Float16* wfc   = (_Float16*)alloc((size_t)Oz * Hz * 2);
    _Float16* xh    = (_Float16*)alloc((size_t)Sz * Bz * Iz * 2);
    _Float16* xp    = (_Float16*)alloc((size_t)Sz * Bz * Gz * 2);  // reused L0/L1
    _Float16* y0    = (_Float16*)alloc((size_t)Sz * Bz * Hz * 2);
    _Float16* hA0   = (_Float16*)alloc((size_t)Bz * Hz * 2);
    _Float16* hA1   = (_Float16*)alloc((size_t)Bz * Hz * 2);
    _Float16* hB0   = (_Float16*)alloc((size_t)Bz * Hz * 2);
    _Float16* hB1   = (_Float16*)alloc((size_t)Bz * Hz * 2);
    _Float16* ybuf  = (_Float16*)alloc((size_t)Bz * 32 * 2);
    float*    cA    = (float*)alloc((size_t)Bz * Hz * 4);
    float*    cB    = (float*)alloc((size_t)Bz * Hz * 4);

    auto cvt = [&](const float* src, _Float16* dst, int rows, int scols, int dcols) {
        int total = rows * dcols;
        cvt_f16_kernel<<<(total + 255) / 256, 256, 0, stream>>>(src, dst, rows, scols, dcols);
    };

    // ---- weight conversion to f16 ----
    cvt(enc_Wih0, wih0e, Gz, Iz, Iz);
    cvt(enc_Whh0, whh0e, Gz, Hz, Hz);
    cvt(enc_Wih1, wih1e, Gz, Hz, Hz);
    cvt(enc_Whh1, whh1e, Gz, Hz, Hz);
    cvt(dec_Wih0, wih0d, Gz, Oz, 32);
    cvt(dec_Whh0, whh0d, Gz, Hz, Hz);
    cvt(dec_Wih1, wih1d, Gz, Hz, Hz);
    cvt(dec_Whh1, whh1d, Gz, Hz, Hz);
    cvt(fc_W,     wfc,   Oz, Hz, Hz);

    // ---- x -> time-major f16, zero states ----
    cvt_x_kernel<<<(Sz * Bz * Iz + 255) / 256, 256, 0, stream>>>(x, xh);
    init_state_kernel<<<(Bz * Hz + 255) / 256, 256, 0, stream>>>(hA0, hB0, cA, cB, ybuf);

    // ---- encoder layer 0: hoisted input GEMM, then recurrence ----
    {
        int waves = (Sz * Bz / 16) * (Gz / 16);
        wmma_gemm_kernel<<<waves * 32 / 256, 256, 0, stream>>>(xh, wih0e, xp, Sz * Bz, Gz, Iz);
    }
    _Float16 *h0c = hA0, *h0n = hA1;
    for (int s = 0; s < Sz; ++s) {
        lstm_step_kernel<<<16, 256, 0, stream>>>(
            h0c, whh0e, Hz, nullptr, nullptr, 0,
            xp + (size_t)s * Bz * Gz, enc_b0, cA, h0n,
            y0 + (size_t)s * Bz * Hz);
        _Float16* t = h0c; h0c = h0n; h0n = t;
    }

    // ---- encoder layer 1 ----
    {
        int waves = (Sz * Bz / 16) * (Gz / 16);
        wmma_gemm_kernel<<<waves * 32 / 256, 256, 0, stream>>>(y0, wih1e, xp, Sz * Bz, Gz, Hz);
    }
    _Float16 *h1c = hB0, *h1n = hB1;
    for (int s = 0; s < Sz; ++s) {
        lstm_step_kernel<<<16, 256, 0, stream>>>(
            h1c, whh1e, Hz, nullptr, nullptr, 0,
            xp + (size_t)s * Bz * Gz, enc_b1, cB, h1n, nullptr);
        _Float16* t = h1c; h1c = h1n; h1n = t;
    }

    // ---- decoder: autoregressive, states continue from encoder finals ----
    for (int s = 0; s < Sz; ++s) {
        // layer 0: inputs = feedback y (K=32 padded) + recurrent h0
        lstm_step_kernel<<<16, 256, 0, stream>>>(
            h0c, whh0d, Hz, ybuf, wih0d, 32,
            nullptr, dec_b0, cA, h0n, nullptr);
        // layer 1: inputs = new h0 + recurrent h1
        lstm_step_kernel<<<16, 256, 0, stream>>>(
            h1c, whh1d, Hz, h0n, wih1d, Hz,
            nullptr, dec_b1, cB, h1n, nullptr);
        // projection + feedback
        fc_step_kernel<<<1, 128, 0, stream>>>(h1n, wfc, fc_b, out, s, ybuf);
        _Float16* t;
        t = h0c; h0c = h0n; h0n = t;
        t = h1c; h1c = h1n; h1n = t;
    }
}
